// NA_28166395527733
// MI455X (gfx1250) — matrix operand order repaired and checked by reading, not compiled, not run on hardware
//
#include <hip/hip_runtime.h>

typedef __attribute__((ext_vector_type(16))) _Float16 v16h;
typedef __attribute__((ext_vector_type(8)))  _Float16 v8h;
typedef __attribute__((ext_vector_type(8)))  float    v8f;

#define HEADS 8
#define HDIM  32
#define HH    64
#define WWID  64
#define KWIN  9
#define HALFW 4
#define NKP   224   // padded candidate-key count: 14 x16 tiles == 7 x32 chunks
#define NT16  14    // logit N-tiles (compile-time -> full unroll)
#define NT32  7     // PV K-chunks  (compile-time -> full unroll)
#define KSTR  40    // sK  row stride (halves): 80B, 16B-aligned, conflict-free
#define VSTR  232   // sVt row stride (halves): 464B, 16B-aligned, conflict-free
#define PSTR  224   // sP  row stride (floats)
#define PHSTR 232   // sPh row stride (halves): 464B, 16B-aligned

__device__ __forceinline__ int iclamp(int x, int lo, int hi) {
  return x < lo ? lo : (x > hi ? hi : x);
}

// 16 contiguous halves from LDS as two 16B (ds_load_b128) chunks
__device__ __forceinline__ v16h lds_ld16h(const _Float16* p) {
  v8h a = *(const v8h*)(p);
  v8h b = *(const v8h*)(p + 8);
  v16h r;
  #pragma unroll
  for (int e = 0; e < 8; ++e) { r[e] = a[e]; r[e + 8] = b[e]; }
  return r;
}

__global__ __launch_bounds__(32) void na2d_wmma_kernel(
    const float* __restrict__ q, const float* __restrict__ k,
    const float* __restrict__ v, float* __restrict__ out)
{
  __shared__ __align__(16) _Float16 sK [NKP * KSTR];   // keys, key-major f16
  __shared__ __align__(16) _Float16 sVt[HDIM * VSTR];  // values, dim-major f16
  __shared__ float                  sP [16 * PSTR];    // f32 logit scoreboard
  __shared__ __align__(16) _Float16 sPh[16 * PHSTR];   // f16 unnormalized probs
  __shared__ float                  sInv[16];          // per-row 1/sum

  const int lane = threadIdx.x;      // wave32
  const int g    = lane >> 4;        // half-wave select
  const int ln   = lane & 15;

  // block -> (head, row i, j-tile)
  const int bid  = blockIdx.x;
  const int head = bid >> 8;
  const int rem  = bid & 255;
  const int i    = rem >> 2;
  const int j0   = (rem & 3) << 4;

  // clamped NATTEN window geometry (uniform across the 16-query tile)
  const int si = iclamp(i - HALFW, 0, HH - KWIN);
  const int cs = iclamp(j0 - HALFW, 0, WWID - KWIN);
  const int ce = iclamp(j0 + 15 - HALFW, 0, WWID - KWIN) + KWIN;
  const int wN = ce - cs;                    // 20 or 24
  const int NK = KWIN * wN;                  // 180 or 216 (<= NKP)

  const size_t hb   = (size_t)head * (HH * WWID * HDIM);
  const size_t base = hb + (size_t)i * (WWID * HDIM);

  // ---- stage K (key-major) and V (dim-major) neighborhoods as f16 ----
  for (int r = 0; r < KWIN; ++r) {
    const size_t rbase = hb + (size_t)(si + r) * (WWID * HDIM)
                            + (size_t)cs * HDIM;
    const float* krow = k + rbase;
    const float* vrow = v + rbase;
    const int rb = r * wN;
    for (int t = lane; t < wN * HDIM; t += 32) {   // coalesced global reads
      const int c = t >> 5, d = t & 31;
      const int kidx = rb + c;
      sK [kidx * KSTR + d]    = (_Float16)krow[t];
      sVt[d * VSTR + kidx]    = (_Float16)vrow[t];
    }
  }
  // zero-pad keys NK..NKP so all fragment loads are unconditional
  for (int t = lane; t < (NKP - NK) * HDIM; t += 32) {
    const int kidx = NK + (t >> 5), d = t & 31;
    sK [kidx * KSTR + d] = (_Float16)0.0f;
    sVt[d * VSTR + kidx] = (_Float16)0.0f;
  }
  // zero prob scoreboard (only window entries get overwritten later)
  for (int t = lane; t < 16 * PHSTR; t += 32) sPh[t] = (_Float16)0.0f;
  __syncthreads();

  // ---- Q fragment: A-matrix 16x32 f16, pre-scaled by 1/sqrt(D) ----
  const float scale = 0.17677669529663687f;  // 1/sqrt(32)
  v16h aq;
  {
    const float* qrow = q + base + (size_t)(j0 + ln) * HDIM;
    #pragma unroll
    for (int e = 0; e < 8; ++e) {              // A-layout K mapping
      aq[e]     = (_Float16)(qrow[g * 16 + e]     * scale);
      aq[e + 8] = (_Float16)(qrow[g * 16 + 8 + e] * scale);
    }
  }

  // ---- logits: Q(16x32) x K^T(32x224) -- 14 straight-line WMMAs ----
  #pragma unroll
  for (int kt = 0; kt < NT16; ++kt) {
    const v16h bk = lds_ld16h(sK + (kt * 16 + ln) * KSTR + g * 16);
    v8f c = {};
    c = __builtin_amdgcn_wmma_f32_16x16x32_f16(false, aq, false, bk,
                                               (short)0, c, false, false);
    #pragma unroll
    for (int r = 0; r < 8; ++r)               // C-layout: M = r + 8g, N = ln
      sP[(r + g * 8) * PSTR + kt * 16 + ln] = c[r];
  }
  __syncthreads();

  // ---- softmax: 2 lanes per query row, exact 81-key window ----
  {
    const int m = lane >> 1, h = lane & 1;
    const int lo = iclamp(j0 + m - HALFW, 0, WWID - KWIN) - cs;
    const float* rowL = sP + m * PSTR;
    _Float16*   rowP = sPh + m * PHSTR;
    float mx = -3.4e38f;
    for (int r = h; r < KWIN; r += 2) {
      const int idx = r * wN + lo;
      #pragma unroll
      for (int c = 0; c < KWIN; ++c) mx = fmaxf(mx, rowL[idx + c]);
    }
    mx = fmaxf(mx, __shfl_xor(mx, 1));
    float sum = 0.0f;
    for (int r = h; r < KWIN; r += 2) {
      const int idx = r * wN + lo;
      #pragma unroll
      for (int c = 0; c < KWIN; ++c) {
        const float e = __expf(rowL[idx + c] - mx);
        sum += e;
        rowP[idx + c] = (_Float16)e;          // unnormalized prob, f16
      }
    }
    sum += __shfl_xor(sum, 1);
    if (h == 0) sInv[m] = 1.0f / sum;         // normalization -> epilogue
  }
  __syncthreads();

  // ---- PV: attn(16x224) x V(224x32) -- 14 straight-line WMMAs ----
  float invs[8];
  #pragma unroll
  for (int r = 0; r < 8; ++r) invs[r] = sInv[g * 8 + r];

  #pragma unroll
  for (int nt = 0; nt < 2; ++nt) {
    v8f acc = {};
    #pragma unroll
    for (int ck = 0; ck < NT32; ++ck) {
      v16h ap;                                 // A-layout: two b128 LDS loads
      {
        const _Float16* p = sPh + ln * PHSTR + ck * 32 + g * 16;
        v8h a0 = *(const v8h*)(p);
        v8h a1 = *(const v8h*)(p + 8);
        #pragma unroll
        for (int e = 0; e < 8; ++e) { ap[e] = a0[e]; ap[e + 8] = a1[e]; }
      }
      const v16h bv =                          // B-layout: two b128 LDS loads
          lds_ld16h(sVt + (nt * 16 + ln) * VSTR + ck * 32 + g * 16);
      acc = __builtin_amdgcn_wmma_f32_16x16x32_f16(false, ap, false, bv,
                                                   (short)0, acc, false, false);
    }
    #pragma unroll
    for (int r = 0; r < 8; ++r)
      out[base + (size_t)(j0 + r + g * 8) * HDIM + nt * 16 + ln]
          = acc[r] * invs[r];
  }
}

extern "C" void kernel_launch(void* const* d_in, const int* in_sizes, int n_in,
                              void* d_out, int out_size, void* d_ws, size_t ws_size,
                              hipStream_t stream) {
  const float* q = (const float*)d_in[0];
  const float* k = (const float*)d_in[1];
  const float* v = (const float*)d_in[2];
  float* out = (float*)d_out;
  // 8 heads * 64 rows * 4 j-tiles = 2048 tiles, one wave32 each
  dim3 grid(HEADS * HH * (WWID / 16));
  na2d_wmma_kernel<<<grid, 32, 0, stream>>>(q, k, v, out);
}
